// TinyFashionNetPureFISUltraConfigurable_3719441679087
// MI455X (gfx1250) — matrix operand order repaired and checked by reading, not compiled, not run on hardware
//
#include <hip/hip_runtime.h>
#include <hip/hip_bf16.h>
#include <math.h>

typedef __attribute__((ext_vector_type(16))) __bf16 v16bf;
typedef __attribute__((ext_vector_type(8)))  float  v8f;

#define LN_EPS 1e-5f

constexpr int B  = 32, H = 128, W = 128;
constexpr int C0 = 32, C1 = 64, NC = 10;
constexpr int H2 = 64, W2 = 64;
constexpr int NPIX1 = B * H * W;    // 524288
constexpr int NPIX2 = B * H2 * W2;  // 131072

// ---------------------------------------------------------------------------
// Convert conv weights wc[co][ci][ky][kx] (f32) -> wbf[tap][co][ci] (bf16)
// ---------------------------------------------------------------------------
__global__ void k_wcvt(const float* __restrict__ wc, __bf16* __restrict__ wbf) {
  int id = blockIdx.x * 256 + threadIdx.x;
  if (id >= 9 * C1 * C1) return;
  int tap = id / (C1 * C1);
  int rem = id % (C1 * C1);
  int co = rem / C1, ci = rem % C1;
  wbf[id] = (__bf16)wc[(size_t)(co * C1 + ci) * 9 + tap];
}

// ---------------------------------------------------------------------------
// Stage 1+2 fused: conv1x1 (3->32) + LayerNorm(g0,b0) + dual GEMM (32->64)
// per-pixel, via WMMA bf16. Outputs z1,z2 channels-last [pix][64] f32.
// One wave = 16 pixels. Lane builds its A-fragment in contiguous k order (k = 16*hi + e), as B does.
// ---------------------------------------------------------------------------
__global__ __launch_bounds__(256)
void k_stage1(const float* __restrict__ x, const float* __restrict__ w_in,
              const float* __restrict__ g0, const float* __restrict__ b0,
              const float* __restrict__ a11, const float* __restrict__ a21,
              float* __restrict__ z1, float* __restrict__ z2) {
  __shared__ __bf16 sA11[C1 * C0];
  __shared__ __bf16 sA21[C1 * C0];
  __shared__ float  sWin[C0 * 3];
  __shared__ float  sG[C0], sB[C0];
  int tid = threadIdx.x;
  for (int i = tid; i < C1 * C0; i += 256) {
    sA11[i] = (__bf16)a11[i];
    sA21[i] = (__bf16)a21[i];
  }
  for (int i = tid; i < C0 * 3; i += 256) sWin[i] = w_in[i];
  if (tid < C0) { sG[tid] = g0[tid]; sB[tid] = b0[tid]; }
  __syncthreads();

  int wave = tid >> 5, lane = tid & 31;
  int hi = lane >> 4, mrow = lane & 15, nl = lane & 15;
  int base = (blockIdx.x * 8 + wave) * 16;
  int pix  = base + mrow;
  int b    = pix >> 14;       // / (128*128)
  int rem  = pix & 16383;
  int hh   = rem >> 7, ww = rem & 127;

  const float* xp = x + (size_t)b * 3 * (H * W) + hh * W + ww;
  float x0 = xp[0], x1 = xp[H * W], x2 = xp[2 * H * W];

  // conv1x1 for the 16 channels this lane owns (contiguous k order: c = 16*hi + e)
  float v[16];
  int   ch[16];
  float s1 = 0.f, s2 = 0.f;
#pragma unroll
  for (int e = 0; e < 16; ++e) {
    int c = hi * 16 + e;
    ch[e] = c;
    float t = x0 * sWin[c * 3 + 0] + x1 * sWin[c * 3 + 1] + x2 * sWin[c * 3 + 2];
    v[e] = t; s1 += t; s2 += t * t;
  }
  // partner lane (lane ^ 16) holds the other 16 channels of the same pixel
  s1 += __shfl_xor(s1, 16, 32);
  s2 += __shfl_xor(s2, 16, 32);
  float mean = s1 * (1.f / 32.f);
  float var  = s2 * (1.f / 32.f) - mean * mean;
  float rstd = rsqrtf(var + LN_EPS);

  v16bf fa;
#pragma unroll
  for (int e = 0; e < 16; ++e)
    fa[e] = (__bf16)(((v[e] - mean) * rstd) * sG[ch[e]] + sB[ch[e]]);

  int kb = hi * 16;
#pragma unroll
  for (int nt = 0; nt < 4; ++nt) {
    int n = nt * 16 + nl;
    v16bf fb1, fb2;
#pragma unroll
    for (int j = 0; j < 8; ++j) {
      fb1[2 * j]     = sA11[n * C0 + kb + 2 * j];
      fb1[2 * j + 1] = sA11[n * C0 + kb + 2 * j + 1];
      fb2[2 * j]     = sA21[n * C0 + kb + 2 * j];
      fb2[2 * j + 1] = sA21[n * C0 + kb + 2 * j + 1];
    }
    v8f zero = {};
    v8f acc1 = __builtin_amdgcn_wmma_f32_16x16x32_bf16(false, fa, false, fb1,
                                                       (short)0, zero, false, false);
    v8f acc2 = __builtin_amdgcn_wmma_f32_16x16x32_bf16(false, fa, false, fb2,
                                                       (short)0, zero, false, false);
#pragma unroll
    for (int r = 0; r < 8; ++r) {
      int m = r + hi * 8;
      size_t o = (size_t)(base + m) * C1 + nt * 16 + nl;
      z1[o] = acc1[r];
      z2[o] = acc2[r];
    }
  }
}

// ---------------------------------------------------------------------------
// Prefix-max scans (channels-last -> adjacent threads touch adjacent floats)
// ---------------------------------------------------------------------------
__global__ void k_rowscan(float* __restrict__ z, int Bn, int Hn, int Wn) {
  int id = blockIdx.x * 256 + threadIdx.x;
  if (id >= Bn * Hn * C1) return;
  int t = id % C1, rh = id / C1;
  int h = rh % Hn, b = rh / Hn;
  float* p = z + ((size_t)(b * Hn + h) * Wn) * C1 + t;
  float m = -INFINITY;
  for (int w = 0; w < Wn; ++w) {
    size_t o = (size_t)w * C1;
    m = fmaxf(m, p[o]);
    p[o] = m;
  }
}

__global__ void k_colscan(float* __restrict__ z, int Bn, int Hn, int Wn) {
  int id = blockIdx.x * 256 + threadIdx.x;
  if (id >= Bn * Wn * C1) return;
  int t = id % C1, rw = id / C1;
  int w = rw % Wn, b = rw / Wn;
  float* p = z + ((size_t)b * Hn * Wn + w) * C1 + t;
  float m = -INFINITY;
  for (int h = 0; h < Hn; ++h) {
    size_t o = (size_t)h * Wn * C1;
    m = fmaxf(m, p[o]);
    p[o] = m;
  }
}

// ---------------------------------------------------------------------------
// z = LayerNorm(z + zc) over 64 channels (in place on z). One wave per pixel.
// ---------------------------------------------------------------------------
__global__ __launch_bounds__(256)
void k_addln(const float* __restrict__ zc, float* __restrict__ z,
             const float* __restrict__ g, const float* __restrict__ bb, int npix) {
  int wave = threadIdx.x >> 5, lane = threadIdx.x & 31;
  int pix = blockIdx.x * 8 + wave;
  if (pix >= npix) return;
  size_t o = (size_t)pix * C1;
  float v0 = z[o + lane]      + zc[o + lane];
  float v1 = z[o + lane + 32] + zc[o + lane + 32];
  float s1 = v0 + v1, s2 = v0 * v0 + v1 * v1;
#pragma unroll
  for (int off = 16; off >= 1; off >>= 1) {
    s1 += __shfl_xor(s1, off, 32);
    s2 += __shfl_xor(s2, off, 32);
  }
  float mean = s1 * (1.f / 64.f);
  float var  = s2 * (1.f / 64.f) - mean * mean;
  float rstd = rsqrtf(var + LN_EPS);
  z[o + lane]      = (v0 - mean) * rstd * g[lane]      + bb[lane];
  z[o + lane + 32] = (v1 - mean) * rstd * g[lane + 32] + bb[lane + 32];
}

// ---------------------------------------------------------------------------
// 2x2 max pool (channels-last, elementwise over t)
// ---------------------------------------------------------------------------
__global__ void k_pool(const float* __restrict__ z, float* __restrict__ p) {
  int id = blockIdx.x * 256 + threadIdx.x;    // B*64*64*64 threads
  int t = id & 63, j = (id >> 6) & 63, i = (id >> 12) & 63, b = id >> 18;
  size_t r0 = ((size_t)((b * H + 2 * i) * W + 2 * j)) * C1 + t;
  size_t r1 = r0 + (size_t)W * C1;
  p[id] = fmaxf(fmaxf(z[r0], z[r0 + C1]), fmaxf(z[r1], z[r1 + C1]));
}

// ---------------------------------------------------------------------------
// LayerNorm(g2,b2) of pooled activations -> bf16 (conv input)
// ---------------------------------------------------------------------------
__global__ __launch_bounds__(256)
void k_lnbf(const float* __restrict__ p, __bf16* __restrict__ pb,
            const float* __restrict__ g, const float* __restrict__ bb) {
  int wave = threadIdx.x >> 5, lane = threadIdx.x & 31;
  int pix = blockIdx.x * 8 + wave;
  size_t o = (size_t)pix * C1;
  float v0 = p[o + lane], v1 = p[o + lane + 32];
  float s1 = v0 + v1, s2 = v0 * v0 + v1 * v1;
#pragma unroll
  for (int off = 16; off >= 1; off >>= 1) {
    s1 += __shfl_xor(s1, off, 32);
    s2 += __shfl_xor(s2, off, 32);
  }
  float mean = s1 * (1.f / 64.f);
  float var  = s2 * (1.f / 64.f) - mean * mean;
  float rstd = rsqrtf(var + LN_EPS);
  pb[o + lane]      = (__bf16)((v0 - mean) * rstd * g[lane]      + bb[lane]);
  pb[o + lane + 32] = (__bf16)((v1 - mean) * rstd * g[lane + 32] + bb[lane + 32]);
}

// ---------------------------------------------------------------------------
// 3x3 conv 64->64 as implicit GEMM (K = 9 taps x 64 ci), WMMA bf16, ReLU.
// Wave = 16 output pixels; per-tap weight slice staged in LDS.
// ---------------------------------------------------------------------------
__global__ __launch_bounds__(256)
void k_conv(const __bf16* __restrict__ pb, const __bf16* __restrict__ wb,
            float* __restrict__ out) {
  __shared__ __bf16 sW[C1 * C1];   // one tap: [co][ci]
  int tid = threadIdx.x;
  int wave = tid >> 5, lane = tid & 31;
  int hi = lane >> 4, mrow = lane & 15, nl = lane & 15;
  int base = (blockIdx.x * 8 + wave) * 16;
  int pix = base + mrow;
  int b = pix >> 12, rem = pix & 4095;
  int i = rem >> 6, j = rem & 63;

  v8f zero = {};
  v8f acc[4];
#pragma unroll
  for (int nt = 0; nt < 4; ++nt) acc[nt] = zero;

  for (int tap = 0; tap < 9; ++tap) {
    __syncthreads();
    for (int q = tid; q < C1 * C1; q += 256) sW[q] = wb[tap * C1 * C1 + q];
    __syncthreads();

    int dy = tap / 3 - 1, dx = tap % 3 - 1;
    int iy = i + dy, ix = j + dx;
    bool ok = (iy >= 0) && (iy < H2) && (ix >= 0) && (ix < W2);
    const __bf16* ap = pb + ((size_t)((b * H2 + iy) * W2 + ix)) * C1;

#pragma unroll
    for (int ks = 0; ks < 2; ++ks) {
      v16bf fa;
#pragma unroll
      for (int e = 0; e < 16; ++e) {
        int c = ks * 32 + hi * 16 + e;
        fa[e] = ok ? ap[c] : (__bf16)0.0f;
      }
      int kb = ks * 32 + hi * 16;
#pragma unroll
      for (int nt = 0; nt < 4; ++nt) {
        int n = nt * 16 + nl;
        v16bf fb;
#pragma unroll
        for (int jj = 0; jj < 8; ++jj) {
          fb[2 * jj]     = sW[n * C1 + kb + 2 * jj];
          fb[2 * jj + 1] = sW[n * C1 + kb + 2 * jj + 1];
        }
        acc[nt] = __builtin_amdgcn_wmma_f32_16x16x32_bf16(false, fa, false, fb,
                                                          (short)0, acc[nt], false, false);
      }
    }
  }

#pragma unroll
  for (int nt = 0; nt < 4; ++nt)
#pragma unroll
    for (int r = 0; r < 8; ++r) {
      int m = r + hi * 8;
      out[(size_t)(base + m) * C1 + nt * 16 + nl] = fmaxf(acc[nt][r], 0.f);
    }
}

// ---------------------------------------------------------------------------
// Stage 5 fused: LayerNorm(g3,b3) + dual GEMM (64->64) via WMMA bf16.
// ---------------------------------------------------------------------------
__global__ __launch_bounds__(256)
void k_fis2(const float* __restrict__ xin, const float* __restrict__ g,
            const float* __restrict__ bb, const float* __restrict__ a1,
            const float* __restrict__ a2, float* __restrict__ z1,
            float* __restrict__ z2) {
  __shared__ __bf16 sA1[C1 * C1], sA2[C1 * C1];
  __shared__ float  sG[C1], sB[C1];
  int tid = threadIdx.x;
  for (int q = tid; q < C1 * C1; q += 256) {
    sA1[q] = (__bf16)a1[q];
    sA2[q] = (__bf16)a2[q];
  }
  if (tid < C1) { sG[tid] = g[tid]; sB[tid] = bb[tid]; }
  __syncthreads();

  int wave = tid >> 5, lane = tid & 31;
  int hi = lane >> 4, mrow = lane & 15, nl = lane & 15;
  int base = (blockIdx.x * 8 + wave) * 16;
  int pix = base + mrow;
  const float* xp = xin + (size_t)pix * C1;

  float v[2][16];
  int   ch[2][16];
  float s1 = 0.f, s2 = 0.f;
#pragma unroll
  for (int ks = 0; ks < 2; ++ks)
#pragma unroll
    for (int e = 0; e < 16; ++e) {
      int c = ks * 32 + hi * 16 + e;
      ch[ks][e] = c;
      float t = xp[c];
      v[ks][e] = t; s1 += t; s2 += t * t;
    }
  s1 += __shfl_xor(s1, 16, 32);
  s2 += __shfl_xor(s2, 16, 32);
  float mean = s1 * (1.f / 64.f);
  float var  = s2 * (1.f / 64.f) - mean * mean;
  float rstd = rsqrtf(var + LN_EPS);

  v16bf fa[2];
#pragma unroll
  for (int ks = 0; ks < 2; ++ks)
#pragma unroll
    for (int e = 0; e < 16; ++e)
      fa[ks][e] = (__bf16)(((v[ks][e] - mean) * rstd) * sG[ch[ks][e]] + sB[ch[ks][e]]);

#pragma unroll
  for (int nt = 0; nt < 4; ++nt) {
    int n = nt * 16 + nl;
    v8f acc1 = {}, acc2 = {};
#pragma unroll
    for (int ks = 0; ks < 2; ++ks) {
      int kb = ks * 32 + hi * 16;
      v16bf fb1, fb2;
#pragma unroll
      for (int jj = 0; jj < 8; ++jj) {
        fb1[2 * jj]     = sA1[n * C1 + kb + 2 * jj];
        fb1[2 * jj + 1] = sA1[n * C1 + kb + 2 * jj + 1];
        fb2[2 * jj]     = sA2[n * C1 + kb + 2 * jj];
        fb2[2 * jj + 1] = sA2[n * C1 + kb + 2 * jj + 1];
      }
      acc1 = __builtin_amdgcn_wmma_f32_16x16x32_bf16(false, fa[ks], false, fb1,
                                                     (short)0, acc1, false, false);
      acc2 = __builtin_amdgcn_wmma_f32_16x16x32_bf16(false, fa[ks], false, fb2,
                                                     (short)0, acc2, false, false);
    }
#pragma unroll
    for (int r = 0; r < 8; ++r) {
      int m = r + hi * 8;
      size_t o = (size_t)(base + m) * C1 + nt * 16 + nl;
      z1[o] = acc1[r];
      z2[o] = acc2[r];
    }
  }
}

// ---------------------------------------------------------------------------
// Global max over pixels of (z2 + cummax(z1)) -> feat[B][64]
// ---------------------------------------------------------------------------
__device__ inline void atomicMaxFloat(float* addr, float val) {
  unsigned int* ua = (unsigned int*)addr;
  unsigned int old = *ua;
  while (__uint_as_float(old) < val) {
    unsigned int assumed = old;
    old = atomicCAS(ua, assumed, __float_as_uint(val));
    if (old == assumed) break;
  }
}

__global__ void k_featinit(float* __restrict__ feat) {
  int id = blockIdx.x * 256 + threadIdx.x;
  if (id < B * C1) feat[id] = -INFINITY;
}

__global__ void k_gmax(const float* __restrict__ z1, const float* __restrict__ z2,
                       float* __restrict__ feat) {
  int id = blockIdx.x * 256 + threadIdx.x;   // (NPIX2/16)*64 threads
  int t = id & 63;
  int pc = id >> 6;          // 0..8191
  int b = pc >> 8;           // 256 pixel-chunks per image
  int pp = pc & 255;
  size_t basepix = (size_t)b * (H2 * W2) + pp * 16;
  float m = -INFINITY;
  for (int k = 0; k < 16; ++k) {
    size_t o = (basepix + k) * C1 + t;
    m = fmaxf(m, z1[o] + z2[o]);
  }
  atomicMaxFloat(&feat[b * C1 + t], m);
}

// ---------------------------------------------------------------------------
// Head: out[b][n] = feat[b] . w_head[n] + b_head[n]
// ---------------------------------------------------------------------------
__global__ void k_head(const float* __restrict__ feat, const float* __restrict__ wh,
                       const float* __restrict__ bh, float* __restrict__ out) {
  int id = threadIdx.x;
  if (id >= B * NC) return;
  int b = id / NC, n = id % NC;
  float acc = bh[n];
  for (int t = 0; t < C1; ++t) acc += feat[b * C1 + t] * wh[n * C1 + t];
  out[b * NC + n] = acc;
}

// ---------------------------------------------------------------------------
extern "C" void kernel_launch(void* const* d_in, const int* in_sizes, int n_in,
                              void* d_out, int out_size, void* d_ws, size_t ws_size,
                              hipStream_t stream) {
  (void)in_sizes; (void)n_in; (void)out_size; (void)ws_size;
  const float* x      = (const float*)d_in[0];
  const float* w_in   = (const float*)d_in[1];
  const float* g0     = (const float*)d_in[2];
  const float* b0     = (const float*)d_in[3];
  const float* a11    = (const float*)d_in[4];
  const float* a21    = (const float*)d_in[5];
  const float* g1     = (const float*)d_in[6];
  const float* b1     = (const float*)d_in[7];
  const float* g2     = (const float*)d_in[8];
  const float* b2     = (const float*)d_in[9];
  const float* wc     = (const float*)d_in[10];
  const float* g3     = (const float*)d_in[11];
  const float* b3     = (const float*)d_in[12];
  const float* a12    = (const float*)d_in[13];
  const float* a22    = (const float*)d_in[14];
  const float* w_head = (const float*)d_in[15];
  const float* b_head = (const float*)d_in[16];
  float* out = (float*)d_out;

  // workspace layout
  float*  z1   = (float*)d_ws;                       // NPIX1*64 f32 (128 MB)
  float*  z2   = z1 + (size_t)NPIX1 * C1;            // NPIX1*64 f32 (128 MB)
  float*  pool = z2 + (size_t)NPIX1 * C1;            // NPIX2*64 f32 (32 MB)
  __bf16* pbf  = (__bf16*)(pool + (size_t)NPIX2 * C1);   // NPIX2*64 bf16 (16 MB)
  float*  cvo  = (float*)(pbf + (size_t)NPIX2 * C1);     // NPIX2*64 f32 (32 MB)
  __bf16* wbf  = (__bf16*)(cvo + (size_t)NPIX2 * C1);    // 9*64*64 bf16
  float*  feat = (float*)(wbf + 9 * C1 * C1);            // B*64 f32

  k_wcvt   <<<(9 * C1 * C1 + 255) / 256, 256, 0, stream>>>(wc, wbf);
  k_stage1 <<<NPIX1 / 128, 256, 0, stream>>>(x, w_in, g0, b0, a11, a21, z1, z2);
  k_rowscan<<<(B * H * C1) / 256, 256, 0, stream>>>(z1, B, H, W);
  k_colscan<<<(B * W * C1) / 256, 256, 0, stream>>>(z1, B, H, W);
  k_addln  <<<NPIX1 / 8, 256, 0, stream>>>(z1, z2, g1, b1, NPIX1);
  k_pool   <<<(B * H2 * W2 * C1) / 256, 256, 0, stream>>>(z2, pool);
  k_lnbf   <<<NPIX2 / 8, 256, 0, stream>>>(pool, pbf, g2, b2);
  k_conv   <<<NPIX2 / 128, 256, 0, stream>>>(pbf, wbf, cvo);
  k_fis2   <<<NPIX2 / 128, 256, 0, stream>>>(cvo, g3, b3, a12, a22, z1, z2);
  k_rowscan<<<(B * H2 * C1) / 256, 256, 0, stream>>>(z1, B, H2, W2);
  k_colscan<<<(B * W2 * C1) / 256, 256, 0, stream>>>(z1, B, H2, W2);
  k_featinit<<<(B * C1 + 255) / 256, 256, 0, stream>>>(feat);
  k_gmax   <<<((NPIX2 / 16) * C1) / 256, 256, 0, stream>>>(z1, z2, feat);
  k_head   <<<1, 320, 0, stream>>>(feat, w_head, b_head, out);
}
